// NmnistCnn_30236569764070
// MI455X (gfx1250) — matrix operand order repaired and checked by reading, not compiled, not run on hardware
//
#include <hip/hip_runtime.h>

// ---------------------------------------------------------------------------
// SLAYER NMNIST CNN forward on gfx1250 (MI455X).
//  - All 5 linear layers (3 convs + 2 dense) run as implicit-GEMM with
//    V_WMMA_F32_16X16X32_F16:  M = 16 timesteps, N = 16 out-channels,
//    K = Cin*KH*KW.  T is innermost in the tensors, so A-tile rows are
//    16 contiguous floats -> coalesced b128 loads staged through LDS.
//  - Layer geometry is a template parameter: all index math is constant
//    mul/shift, and one A tile in LDS is shared by NT waves (one per
//    16-wide output-channel tile), removing the NT-fold input re-read.
//  - psp+spike IIRs fused into one VALU kernel (float4 I/O), one thread
//    per neuron; 2x2 sum-pool fused into the following IIR kernel.
// ---------------------------------------------------------------------------

typedef __attribute__((ext_vector_type(16))) _Float16 v16h;
typedef __attribute__((ext_vector_type(8)))  float    v8f;

#define T_STEPS 300
#define T_TILES 19                       // ceil(300/16)
#define BATCH   4

// float32 values of the reference IIR constants
#define A_SR     0.90483741803595952f    // exp(-1/10)
#define C_SR     0.27182818284590454f    // e/10
#define A_REF    0.36787944117144233f    // exp(-1)
#define C_REF    2.7182818284590452f     // e
#define THETA    10.0f
#define REF_COEF (2.0f * THETA * C_REF)  // SCALE_REF*theta*c_ref
#define POOL_SCALE 11.0f                 // 1.1*theta

// ---------------------------------------------------------------------------
// Pack f32 weights [Cout][K] into fragment-ready f16 order:
//   wf[((nt*KC + kc)*32 + lane)*16 + i] = W[nt*16 + (lane&15)][kc*32 + i + 16*(lane>>4)]
// so each lane's 16 B-fragment halves are one contiguous 32-byte read.
// (B fragment k order chosen here, matched by the A fragment: lane -> N = lane&15, half i -> K = i + 16*(lane>>4))
// ---------------------------------------------------------------------------
__global__ void pack_weights(const float* __restrict__ w, _Float16* __restrict__ wf,
                             int Cout, int K, int KC, int NT) {
  int idx   = blockIdx.x * blockDim.x + threadIdx.x;
  int total = NT * KC * 512;
  if (idx >= total) return;
  int i    = idx & 15;
  int lane = (idx >> 4) & 31;
  int kc   = (idx >> 9) % KC;
  int nt   = (idx >> 9) / KC;
  int n = nt * 16 + (lane & 15);
  int k = kc * 32 + i + 16 * (lane >> 4);
  float v = 0.0f;
  if (n < Cout && k < K) v = w[(size_t)n * K + k];
  wf[idx] = (_Float16)v;
}

// ---------------------------------------------------------------------------
// Implicit-GEMM conv (also covers dense layers: KH=KW=H=W, pad=0, OH=OW=1).
// Block = (32, NT): NT waves share one staged A tile; wave wy computes the
// 16(t) x 16(co) tile for out-channel tile wy at one output pixel.
//   grid.x = T_TILES * OH * OW, grid.z = batch
// ---------------------------------------------------------------------------
template <int Cin, int H, int W, int Cout, int OH, int OW,
          int KH, int KW, int PAD, int K, int KC, int NT>
__global__ __launch_bounds__(32 * NT)
void conv_wmma(const float* __restrict__ x, const _Float16* __restrict__ wf,
               float* __restrict__ y) {
  const int lane = threadIdx.x;
  const int nt   = threadIdx.y;
  const int tt   = blockIdx.x % T_TILES;
  const int pix  = blockIdx.x / T_TILES;
  const int oy   = pix / OW;
  const int ox   = pix % OW;
  const int b    = blockIdx.z;
  const int t0   = tt * 16;

  __shared__ alignas(32) _Float16 lds[32 * 16];   // 32 k-rows x 16 timesteps

  v8f acc = {};

  for (int kc = 0; kc < KC; ++kc) {
    // ---- stage A tile (wave 0 only): lane = k-row, 16 contiguous timesteps ----
    if (nt == 0) {
      float vals[16];
      #pragma unroll
      for (int j = 0; j < 16; ++j) vals[j] = 0.0f;

      int k = kc * 32 + lane;
      if (k < K) {
        int ci = k / (KH * KW);
        int r  = k - ci * (KH * KW);
        int ky = r / KW;
        int kx = r - ky * KW;
        int iy = oy + ky - PAD;
        int ix = ox + kx - PAD;
        if (iy >= 0 && iy < H && ix >= 0 && ix < W) {
          const float* src = x +
              ((((size_t)b * Cin + ci) * H + iy) * W + ix) * (size_t)T_STEPS + t0;
          if (t0 + 16 <= T_STEPS) {              // 300*4 and 64*tt are 16B multiples
            const float4* s4 = (const float4*)src;
            #pragma unroll
            for (int j = 0; j < 4; ++j) {
              float4 v = s4[j];
              vals[4*j+0] = v.x; vals[4*j+1] = v.y; vals[4*j+2] = v.z; vals[4*j+3] = v.w;
            }
          } else {                               // last partial time tile
            #pragma unroll
            for (int j = 0; j < 16; ++j) if (t0 + j < T_STEPS) vals[j] = src[j];
          }
        }
      }
      v16h hv;
      #pragma unroll
      for (int j = 0; j < 16; ++j) hv[j] = (_Float16)vals[j];
      *(v16h*)(&lds[lane * 16]) = hv;
    }
    __syncthreads();

    // ---- A fragment (matches packed B): half i -> K=i+16*g ----
    const int g = lane >> 4;
    const int m = lane & 15;
    v16h afrag;
    #pragma unroll
    for (int i = 0; i < 16; ++i) {
      int kk = i + 16 * g;
      afrag[i] = lds[kk * 16 + m];
    }

    // ---- B fragment: pre-packed, contiguous 32B per lane ----
    const _Float16* wsrc = wf + (((size_t)nt * KC + kc) * 32 + lane) * 16;
    v16h bfrag = *(const v16h*)wsrc;
    if (kc + 1 < KC) __builtin_prefetch(wsrc + 512, 0, 1);   // next K chunk

    acc = __builtin_amdgcn_wmma_f32_16x16x32_f16(false, afrag, false, bfrag,
                                                 (short)0, acc, false, false);
    __syncthreads();
  }

  // ---- store D: vgpr r, lane -> (t = t0 + r + 8*g, co = nt*16 + (lane&15)) ----
  const int g = lane >> 4;
  const int n = nt * 16 + (lane & 15);
  if (n < Cout) {
    float* dst = y + ((((size_t)b * Cout + n) * OH + oy) * OW + ox) * (size_t)T_STEPS;
    #pragma unroll
    for (int r = 0; r < 8; ++r) {
      int t = t0 + r + 8 * g;
      if (t < T_STEPS) dst[t] = acc[r];
    }
  }
}

// ---------------------------------------------------------------------------
// Fused psp (alpha-kernel IIR) + spike (threshold w/ refractory IIR).
// One thread per neuron row of T contiguous samples; float4 I/O.
// In-place safe (each float4 written after it is read).
// ---------------------------------------------------------------------------
__device__ __forceinline__ float iir_step(float xin, float& p1, float& q1,
                                          float& p2, float& q2) {
  q1 = A_SR * (q1 + p1);            // psp: q uses old p
  p1 = A_SR * p1 + xin;
  float ut = C_SR * q1;
  q2 = A_REF * (q2 + p2);           // refractory: q uses old p
  float um = ut - REF_COEF * q2;
  float sp = (um >= THETA) ? 1.0f : 0.0f;
  p2 = A_REF * p2 + sp;
  return sp;
}

__global__ void psp_spike(const float* __restrict__ u, float* __restrict__ s, int nrows) {
  int row = blockIdx.x * blockDim.x + threadIdx.x;
  if (row >= nrows) return;
  const float4* in4  = (const float4*)(u + (size_t)row * T_STEPS);
  float4*       out4 = (float4*)(s + (size_t)row * T_STEPS);
  float p1 = 0.f, q1 = 0.f, p2 = 0.f, q2 = 0.f;
  for (int tb = 0; tb < T_STEPS / 4; ++tb) {
    float4 xv = in4[tb];
    float4 sv;
    sv.x = iir_step(xv.x, p1, q1, p2, q2);
    sv.y = iir_step(xv.y, p1, q1, p2, q2);
    sv.z = iir_step(xv.z, p1, q1, p2, q2);
    sv.w = iir_step(xv.w, p1, q1, p2, q2);
    out4[tb] = sv;
  }
}

// ---------------------------------------------------------------------------
// Fused 2x2 sum-pool (*1.1*theta, zero-pad odd dims) + psp + spike.
// ---------------------------------------------------------------------------
__global__ void pool_psp_spike(const float* __restrict__ x, float* __restrict__ s,
                               int C, int H, int W, int OH, int OW, int nrows) {
  int idx = blockIdx.x * blockDim.x + threadIdx.x;
  if (idx >= nrows) return;
  int ox = idx % OW;
  int oy = (idx / OW) % OH;
  int c  = (idx / (OW * OH)) % C;
  int b  =  idx / (OW * OH * C);

  const float4* rows[4];
  int nv = 0;
  #pragma unroll
  for (int dy = 0; dy < 2; ++dy)
    #pragma unroll
    for (int dx = 0; dx < 2; ++dx) {
      int iy = 2 * oy + dy, ix = 2 * ox + dx;
      if (iy < H && ix < W)
        rows[nv++] = (const float4*)(x +
            ((((size_t)b * C + c) * H + iy) * W + ix) * (size_t)T_STEPS);
    }
  float4* out4 = (float4*)(s + (size_t)idx * T_STEPS);

  float p1 = 0.f, q1 = 0.f, p2 = 0.f, q2 = 0.f;
  for (int tb = 0; tb < T_STEPS / 4; ++tb) {
    float4 xv = {0.f, 0.f, 0.f, 0.f};
    for (int j = 0; j < nv; ++j) {
      float4 v = rows[j][tb];
      xv.x += v.x; xv.y += v.y; xv.z += v.z; xv.w += v.w;
    }
    float4 sv;
    sv.x = iir_step(POOL_SCALE * xv.x, p1, q1, p2, q2);
    sv.y = iir_step(POOL_SCALE * xv.y, p1, q1, p2, q2);
    sv.z = iir_step(POOL_SCALE * xv.z, p1, q1, p2, q2);
    sv.w = iir_step(POOL_SCALE * xv.w, p1, q1, p2, q2);
    out4[tb] = sv;
  }
}

// ---------------------------------------------------------------------------
extern "C" void kernel_launch(void* const* d_in, const int* in_sizes, int n_in,
                              void* d_out, int out_size, void* d_ws, size_t ws_size,
                              hipStream_t stream) {
  const float* s_in = (const float*)d_in[0];   // (4,2,34,34,300)
  const float* Wc1  = (const float*)d_in[1];   // (24,2,5,5)
  const float* Wc2  = (const float*)d_in[2];   // (48,24,3,3)
  const float* Wc3  = (const float*)d_in[3];   // (96,48,3,3)
  const float* Wd4a = (const float*)d_in[4];   // (256,96,5,5)
  const float* Wd4b = (const float*)d_in[5];   // (10,256)
  float* out = (float*)d_out;                  // (4,10,300)

  // ---- workspace carve-up ----
  char* ws = (char*)d_ws;
  size_t off = 0;
  auto carve = [&](size_t bytes) -> char* {
    off = (off + 255) & ~(size_t)255;
    char* p = ws + off;
    off += bytes;
    return p;
  };
  float* bufA = (float*)carve((size_t)4 * 24 * 34 * 34 * 300 * 4);  // 133.2 MB  (conv outs)
  float* bufB = (float*)carve((size_t)4 * 24 * 17 * 17 * 300 * 4);  //  33.3 MB  (pool outs)
  float* bufG = (float*)carve((size_t)4 * 256 * 300 * 4);           //   1.2 MB  (dense4a)
  float* bufH = (float*)carve((size_t)4 * 10 * 300 * 4);            //   48 KB   (dense4b)
  _Float16* wf1  = (_Float16*)carve((size_t)2  * 2  * 512 * 2);
  _Float16* wf2  = (_Float16*)carve((size_t)3  * 7  * 512 * 2);
  _Float16* wf3  = (_Float16*)carve((size_t)6  * 14 * 512 * 2);
  _Float16* wf4a = (_Float16*)carve((size_t)16 * 75 * 512 * 2);
  _Float16* wf4b = (_Float16*)carve((size_t)1  * 8  * 512 * 2);

  auto pack = [&](const float* w, _Float16* wf, int Cout, int K, int KC, int NT) {
    int total = NT * KC * 512;
    pack_weights<<<(total + 255) / 256, 256, 0, stream>>>(w, wf, Cout, K, KC, NT);
  };
  pack(Wc1,  wf1,  24,   50,  2,  2);
  pack(Wc2,  wf2,  48,  216,  7,  3);
  pack(Wc3,  wf3,  96,  432, 14,  6);
  pack(Wd4a, wf4a, 256, 2400, 75, 16);
  pack(Wd4b, wf4b, 10,   256,  8,  1);

  auto iir = [&](const float* u, float* s, int nrows) {
    psp_spike<<<(nrows + 255) / 256, 256, 0, stream>>>(u, s, nrows);
  };
  auto pool_iir = [&](const float* x, float* s, int C, int H, int W, int OH, int OW) {
    int nrows = BATCH * C * OH * OW;
    pool_psp_spike<<<(nrows + 255) / 256, 256, 0, stream>>>(x, s, C, H, W, OH, OW, nrows);
  };

  // L1: conv 5x5 pad2 (2->24, 34x34) -> psp+spike -> pool+psp+spike
  conv_wmma<2, 34, 34, 24, 34, 34, 5, 5, 2, 50, 2, 2>
      <<<dim3(T_TILES * 34 * 34, 1, BATCH), dim3(32, 2), 0, stream>>>(s_in, wf1, bufA);
  iir(bufA, bufA, 4 * 24 * 34 * 34);
  pool_iir(bufA, bufB, 24, 34, 34, 17, 17);

  // L2: conv 3x3 pad1 (24->48, 17x17)
  conv_wmma<24, 17, 17, 48, 17, 17, 3, 3, 1, 216, 7, 3>
      <<<dim3(T_TILES * 17 * 17, 1, BATCH), dim3(32, 3), 0, stream>>>(bufB, wf2, bufA);
  iir(bufA, bufA, 4 * 48 * 17 * 17);
  pool_iir(bufA, bufB, 48, 17, 17, 9, 9);

  // L3: conv 3x3 pad1 (48->96, 9x9)
  conv_wmma<48, 9, 9, 96, 9, 9, 3, 3, 1, 432, 14, 6>
      <<<dim3(T_TILES * 9 * 9, 1, BATCH), dim3(32, 6), 0, stream>>>(bufB, wf3, bufA);
  iir(bufA, bufA, 4 * 96 * 9 * 9);
  pool_iir(bufA, bufB, 96, 9, 9, 5, 5);

  // Dense 4a: = conv 5x5 valid on 5x5 image (96->256)
  conv_wmma<96, 5, 5, 256, 1, 1, 5, 5, 0, 2400, 75, 16>
      <<<dim3(T_TILES, 1, BATCH), dim3(32, 16), 0, stream>>>(bufB, wf4a, bufG);
  iir(bufG, bufG, 4 * 256);

  // Dense 4b: = conv 1x1 on 1x1 image (256->10); final psp+spike -> d_out
  conv_wmma<256, 1, 1, 10, 1, 1, 1, 1, 0, 256, 8, 1>
      <<<dim3(T_TILES, 1, BATCH), dim3(32, 1), 0, stream>>>(bufG, wf4b, bufH);
  iir(bufH, out, 4 * 10);
}
